// MultiHeadAttention_21981642621504
// MI455X (gfx1250) — matrix operand order repaired and checked by reading, not compiled, not run on hardware
//
#include <hip/hip_runtime.h>

// ---------------------------------------------------------------------------
// MultiHeadAttention forward for MI455X (gfx1250, wave32, WMMA).
//  B=2, S=4096, D=768, H=12, Dh=64.
//  Pipeline:
//   0) one-time f32->bf16 conversion of activations and weights
//   1) QKV projection GEMMs (bf16 WMMA, 32x64 wave tiles, head-split output)
//   2) flash attention (async global->LDS K tiles, double-buffered, online
//      softmax, all matmuls on v_wmma_f32_16x16x32_bf16)
//   3) output projection GEMM to fp32
// ---------------------------------------------------------------------------

typedef __bf16 bf16;
typedef __attribute__((ext_vector_type(16))) __bf16 v16bf;
typedef __attribute__((ext_vector_type(8)))  __bf16 v8bf;
typedef __attribute__((ext_vector_type(8)))  float  v8f;

#define WMMA_BF16(a, b, c) \
  __builtin_amdgcn_wmma_f32_16x16x32_bf16(false, (a), false, (b), (short)0, (c), false, false)

#define B_BATCH 2
#define S_LEN   4096
#define D_MODEL 768
#define HEADS   12
#define DH      64
#define NT      (S_LEN / 32)

// LDS strides (bf16 elements): stride*2 % 16 == 0 (16B-aligned v8bf access)
// and 16-lane row-walks land on 16 distinct 4-bank groups (conflict-free).
#define KT_STRIDE 88
#define VT_STRIDE 40
#define P_STRIDE  56

// --------------------------- fragment loaders ------------------------------
// A 16x32 bf16: lane L row M=L%16; elem j -> K = koff16+j (contiguous,
// same as B).  B 32x16: lane L col N=L%16; elem j ->
// K = koff16+j, koff16=(L>=16)*16.

__device__ __forceinline__ v16bf combine8(v8bf lo, v8bf hi) {
  v16bf r;
#pragma unroll
  for (int i = 0; i < 8; ++i) { r[i] = lo[i]; r[i + 8] = hi[i]; }
  return r;
}

__device__ __forceinline__ v16bf a_frag_bf16(const bf16* row, int kbase, int lane) {
  const int koff = (lane & 16) ? 16 : 0;
  v8bf lo = *(const v8bf*)(row + kbase + koff);
  v8bf hi = *(const v8bf*)(row + kbase + 8 + koff);
  return combine8(lo, hi);
}

__device__ __forceinline__ v16bf b_frag_bf16(const bf16* row, int kbase, int lane) {
  const int koff = (lane & 16) ? 16 : 0;
  const bf16* p = row + kbase + koff;
  v8bf lo = *(const v8bf*)p;
  v8bf hi = *(const v8bf*)(p + 8);
  return combine8(lo, hi);
}

__device__ __forceinline__ v8f v8f_zero() {
  v8f z = {0.f, 0.f, 0.f, 0.f, 0.f, 0.f, 0.f, 0.f};
  return z;
}

// Async global->LDS: moves 32 bytes (2 x b128) per lane; INST_OFFSET is added
// to both the global and LDS addresses (ISA 10, async load pseudocode), so the
// same operands serve both halves. Tracked by ASYNCcnt.
__device__ __forceinline__ void async_load_32B(unsigned long long gbase,
                                               unsigned goff, unsigned loff) {
  asm volatile(
      "global_load_async_to_lds_b128 %0, %1, %2\n\t"
      "global_load_async_to_lds_b128 %0, %1, %2 offset:16"
      :
      : "v"(loff), "v"(goff), "s"(gbase)
      : "memory");
}

__device__ __forceinline__ void wait_async0() {
  asm volatile("s_wait_asynccnt 0" ::: "memory");
}

// --------------------- kernel 0: f32 -> bf16 conversion --------------------
__global__ __launch_bounds__(256) void cvt_bf16_kernel(
    const float* __restrict__ in, bf16* __restrict__ out, int n8) {
  const int i = blockIdx.x * 256 + threadIdx.x;
  if (i >= n8) return;
  const float4* p = (const float4*)(in + (size_t)i * 8);
  float4 a = p[0], b = p[1];
  v8bf o;
  o[0] = (bf16)a.x; o[1] = (bf16)a.y; o[2] = (bf16)a.z; o[3] = (bf16)a.w;
  o[4] = (bf16)b.x; o[5] = (bf16)b.y; o[6] = (bf16)b.z; o[7] = (bf16)b.w;
  *(v8bf*)(out + (size_t)i * 8) = o;
}

// ------------------------- kernel 1: QKV projection ------------------------
// out(bf16, head-split [B,H,S,Dh]) = X @ W^T + bias; Q scaled by 1/sqrt(Dh).
// Grid: (M/128, D/64, 3). Block: 128 (4 waves). Wave tile: 32(M) x 64(N).
__global__ __launch_bounds__(128) void qkv_proj_kernel(
    const bf16* __restrict__ xq, const bf16* __restrict__ xk,
    const bf16* __restrict__ xv,
    const bf16* __restrict__ Wqb, const float* __restrict__ bq,
    const bf16* __restrict__ Wkb, const float* __restrict__ bk,
    const bf16* __restrict__ Wvb, const float* __restrict__ bv,
    bf16* __restrict__ qh, bf16* __restrict__ kh, bf16* __restrict__ vh) {
  const int which = blockIdx.z;
  const bf16* X     = (which == 0) ? xq  : (which == 1) ? xk  : xv;
  const bf16* W     = (which == 0) ? Wqb : (which == 1) ? Wkb : Wvb;
  const float* bias = (which == 0) ? bq  : (which == 1) ? bk  : bv;
  bf16* out         = (which == 0) ? qh  : (which == 1) ? kh  : vh;
  const float scale = (which == 0) ? 0.125f : 1.0f;  // 1/sqrt(64)

  const int tid = threadIdx.x;
  const int lane = tid & 31, wave = tid >> 5;
  const int m0 = blockIdx.x * 128 + wave * 32;
  const int n0 = blockIdx.y * 64;
  const int ml = lane & 15;

  const bf16* xrow0 = X + (size_t)(m0 + ml) * D_MODEL;
  const bf16* xrow1 = X + (size_t)(m0 + 16 + ml) * D_MODEL;

  v8f c[2][4];
#pragma unroll
  for (int mi = 0; mi < 2; ++mi)
#pragma unroll
    for (int f = 0; f < 4; ++f) c[mi][f] = v8f_zero();

  for (int kk = 0; kk < D_MODEL; kk += 32) {
    v16bf a0 = a_frag_bf16(xrow0, kk, lane);
    v16bf a1 = a_frag_bf16(xrow1, kk, lane);
#pragma unroll
    for (int f = 0; f < 4; ++f) {
      const bf16* wrow = W + (size_t)(n0 + f * 16 + ml) * D_MODEL;
      v16bf b = b_frag_bf16(wrow, kk, lane);
      c[0][f] = WMMA_BF16(a0, b, c[0][f]);
      c[1][f] = WMMA_BF16(a1, b, c[1][f]);
    }
  }

  const int rbase = (lane & 16) ? 8 : 0;
#pragma unroll
  for (int f = 0; f < 4; ++f) {
    const int n = n0 + f * 16 + ml;
    const int h = n >> 6, d = n & 63;
    const float bb = bias[n];
#pragma unroll
    for (int mi = 0; mi < 2; ++mi)
#pragma unroll
      for (int r = 0; r < 8; ++r) {
        const int mm = m0 + mi * 16 + r + rbase;
        const int bb_ = mm >> 12, ss = mm & (S_LEN - 1);
        out[(((size_t)bb_ * HEADS + h) * S_LEN + ss) * DH + d] =
            (bf16)((c[mi][f][r] + bb) * scale);
      }
  }
}

// ------------------------- kernel 2: flash attention -----------------------
// Grid: (S/64, B*H). Block: 128 (4 waves); wave owns a 16-row Q strip.
// K tiles go global->LDS via async (ASYNCcnt), V via register transpose;
// both double-buffered so tile t+1 loads overlap tile t compute.
__global__ __launch_bounds__(128) void flash_attn_kernel(
    const bf16* __restrict__ qh, const bf16* __restrict__ kh,
    const bf16* __restrict__ vh, bf16* __restrict__ ctx) {
  __shared__ bf16 ktile[2 * 32 * KT_STRIDE];  // [buf][key][d]
  __shared__ bf16 vt[2 * DH * VT_STRIDE];     // [buf][d][key] (transposed)
  __shared__ bf16 pst[4 * 16 * P_STRIDE];     // per-wave P staging [row][key]

  const int tid = threadIdx.x;
  const int lane = tid & 31, wave = tid >> 5;
  const int ml = lane & 15;
  const int bh = blockIdx.y;
  const int q0 = blockIdx.x * 64 + wave * 16;

  const bf16* Q = qh + (size_t)bh * S_LEN * DH;
  const bf16* K = kh + (size_t)bh * S_LEN * DH;
  const bf16* V = vh + (size_t)bh * S_LEN * DH;

  const bf16* qrow = Q + (size_t)(q0 + ml) * DH;
  v16bf aq0 = a_frag_bf16(qrow, 0, lane);
  v16bf aq1 = a_frag_bf16(qrow, 32, lane);

  v8f acc[4];
#pragma unroll
  for (int dg = 0; dg < 4; ++dg) acc[dg] = v8f_zero();
  float mrow[8], lrow[8];
#pragma unroll
  for (int r = 0; r < 8; ++r) { mrow[r] = -1e30f; lrow[r] = 0.f; }

  bf16* myp = pst + wave * 16 * P_STRIDE;
  const int rbase = (lane & 16) ? 8 : 0;

  // cooperative-load geometry: 128 threads move a 32x64 tile, 32B each.
  const int rr = tid >> 2;          // key row 0..31
  const int cc = (tid & 3) * 16;    // d offset 0,16,32,48
  const unsigned ktbase = (unsigned)(size_t)(void*)ktile;
  const unsigned long long kgb = (unsigned long long)(size_t)K;

  // ---- prologue: tile 0 in flight ----
  async_load_32B(kgb, (unsigned)((rr * DH + cc) * 2),
                 ktbase + (unsigned)((rr * KT_STRIDE + cc) * 2));
  const bf16* gv0 = V + (size_t)rr * DH + cc;
  v8bf va = *(const v8bf*)gv0;
  v8bf vb = *(const v8bf*)(gv0 + 8);

  for (int t = 0; t < NT; ++t) {
    const int p = t & 1;
    bf16* vbuf = vt + p * DH * VT_STRIDE;
    const bf16* kbuf = ktile + p * 32 * KT_STRIDE;

    // stage V tile t (transposed) into buffer p
#pragma unroll
    for (int j = 0; j < 8; ++j) vbuf[(cc + j) * VT_STRIDE + rr] = va[j];
#pragma unroll
    for (int j = 0; j < 8; ++j) vbuf[(cc + 8 + j) * VT_STRIDE + rr] = vb[j];

    wait_async0();     // own K-async portion for tile t complete
    __syncthreads();   // everyone's K portions + V stores visible

    // prefetch tile t+1 into the other buffer (overlaps compute below)
    if (t + 1 < NT) {
      const int tn = t + 1;
      async_load_32B(kgb, (unsigned)(((tn * 32 + rr) * DH + cc) * 2),
                     ktbase + (unsigned)((((tn & 1) * 32 + rr) * KT_STRIDE + cc) * 2));
      const bf16* gv = V + (size_t)(tn * 32 + rr) * DH + cc;
      va = *(const v8bf*)gv;
      vb = *(const v8bf*)(gv + 8);
    }

    // ---- scores: S[16q x 32k] = Q . K^T (Q pre-scaled by 1/8) ----
    v8f s[2];
#pragma unroll
    for (int nf = 0; nf < 2; ++nf) {
      const bf16* krow = kbuf + (nf * 16 + ml) * KT_STRIDE;
      v16bf b0 = b_frag_bf16(krow, 0, lane);
      v16bf b1 = b_frag_bf16(krow, 32, lane);
      v8f z = v8f_zero();
      z = WMMA_BF16(aq0, b0, z);
      s[nf] = WMMA_BF16(aq1, b1, z);
    }

    // ---- online softmax (row reductions within 16-lane half-groups) ----
    float alpha[8];
#pragma unroll
    for (int r = 0; r < 8; ++r) {
      float mx = fmaxf(s[0][r], s[1][r]);
#pragma unroll
      for (int d = 8; d >= 1; d >>= 1) mx = fmaxf(mx, __shfl_xor(mx, d, 32));
      const float mnew = fmaxf(mrow[r], mx);
      const float p0 = __expf(s[0][r] - mnew);
      const float p1 = __expf(s[1][r] - mnew);
      float sum = p0 + p1;
#pragma unroll
      for (int d = 8; d >= 1; d >>= 1) sum += __shfl_xor(sum, d, 32);
      const float al = __expf(mrow[r] - mnew);
      lrow[r] = lrow[r] * al + sum;
      mrow[r] = mnew;
      alpha[r] = al;
      s[0][r] = p0;
      s[1][r] = p1;
    }

    // ---- P: C-layout -> A-layout via wave-private LDS staging ----
#pragma unroll
    for (int nf = 0; nf < 2; ++nf)
#pragma unroll
      for (int r = 0; r < 8; ++r)
        myp[(r + rbase) * P_STRIDE + nf * 16 + ml] = (bf16)s[nf][r];

#pragma unroll
    for (int dg = 0; dg < 4; ++dg)
#pragma unroll
      for (int r = 0; r < 8; ++r) acc[dg][r] *= alpha[r];

    v16bf ap = a_frag_bf16(myp + ml * P_STRIDE, 0, lane);

    // ---- ctx += P . V (B operand from transposed V tile) ----
#pragma unroll
    for (int dg = 0; dg < 4; ++dg) {
      const bf16* vrow = vbuf + (dg * 16 + ml) * VT_STRIDE;
      v16bf bv = b_frag_bf16(vrow, 0, lane);
      acc[dg] = WMMA_BF16(ap, bv, acc[dg]);
    }
  }

  // ---- epilogue: normalize, store ctx bf16 in [B,S,D] layout ----
  float inv[8];
#pragma unroll
  for (int r = 0; r < 8; ++r) inv[r] = 1.0f / lrow[r];
  const int b = bh / HEADS, h = bh % HEADS;
#pragma unroll
  for (int dg = 0; dg < 4; ++dg)
#pragma unroll
    for (int r = 0; r < 8; ++r) {
      const int srow = q0 + r + rbase;
      const int d = dg * 16 + ml;
      ctx[((size_t)(b * S_LEN + srow)) * D_MODEL + h * DH + d] =
          (bf16)(acc[dg][r] * inv[r]);
    }
}

// ------------------------- kernel 3: output projection ---------------------
// out(f32 [B*S, D]) = ctx(bf16) @ Wo^T + bo.  Wave tile 32x64.
__global__ __launch_bounds__(128) void out_proj_kernel(
    const bf16* __restrict__ ctx, const bf16* __restrict__ Wob,
    const float* __restrict__ bo, float* __restrict__ out) {
  const int tid = threadIdx.x;
  const int lane = tid & 31, wave = tid >> 5;
  const int m0 = blockIdx.x * 128 + wave * 32;
  const int n0 = blockIdx.y * 64;
  const int ml = lane & 15;

  const bf16* arow0 = ctx + (size_t)(m0 + ml) * D_MODEL;
  const bf16* arow1 = ctx + (size_t)(m0 + 16 + ml) * D_MODEL;

  v8f c[2][4];
#pragma unroll
  for (int mi = 0; mi < 2; ++mi)
#pragma unroll
    for (int f = 0; f < 4; ++f) c[mi][f] = v8f_zero();

  for (int kk = 0; kk < D_MODEL; kk += 32) {
    v16bf a0 = a_frag_bf16(arow0, kk, lane);
    v16bf a1 = a_frag_bf16(arow1, kk, lane);
#pragma unroll
    for (int f = 0; f < 4; ++f) {
      const bf16* wrow = Wob + (size_t)(n0 + f * 16 + ml) * D_MODEL;
      v16bf b = b_frag_bf16(wrow, kk, lane);
      c[0][f] = WMMA_BF16(a0, b, c[0][f]);
      c[1][f] = WMMA_BF16(a1, b, c[1][f]);
    }
  }

  const int rbase = (lane & 16) ? 8 : 0;
#pragma unroll
  for (int f = 0; f < 4; ++f) {
    const int n = n0 + f * 16 + ml;
    const float bb = bo[n];
#pragma unroll
    for (int mi = 0; mi < 2; ++mi)
#pragma unroll
      for (int r = 0; r < 8; ++r) {
        const int mm = m0 + mi * 16 + r + rbase;
        out[(size_t)mm * D_MODEL + n] = c[mi][f][r] + bb;
      }
  }
}

// ------------------------------- launcher ----------------------------------
extern "C" void kernel_launch(void* const* d_in, const int* in_sizes, int n_in,
                              void* d_out, int out_size, void* d_ws, size_t ws_size,
                              hipStream_t stream) {
  const float* q  = (const float*)d_in[0];
  const float* k  = (const float*)d_in[1];
  const float* v  = (const float*)d_in[2];
  const float* Wq = (const float*)d_in[3];
  const float* bq = (const float*)d_in[4];
  const float* Wk = (const float*)d_in[5];
  const float* bk = (const float*)d_in[6];
  const float* Wv = (const float*)d_in[7];
  const float* bv = (const float*)d_in[8];
  const float* Wo = (const float*)d_in[9];
  const float* bo = (const float*)d_in[10];
  float* out = (float*)d_out;

  const size_t R  = (size_t)B_BATCH * HEADS * S_LEN * DH;  // 6,291,456
  const size_t WE = (size_t)D_MODEL * D_MODEL;             //   589,824

  bf16* qh  = (bf16*)d_ws;
  bf16* kh  = qh + R;
  bf16* vh  = kh + R;
  bf16* ctx = vh + R;
  bf16* xqb = ctx + R;
  bf16* xkb = xqb + R;
  bf16* xvb = xkb + R;
  bf16* wqb = xvb + R;
  bf16* wkb = wqb + WE;
  bf16* wvb = wkb + WE;
  bf16* wob = wvb + WE;   // total (7R + 4WE)*2 bytes ~= 89 MB

  const int M = B_BATCH * S_LEN;  // 8192

  // 0) one-time bf16 conversions (HBM-bound, ~8us total)
  const int n8x = (int)(R / 8), n8w = (int)(WE / 8);
  const int gx = (n8x + 255) / 256, gw = (n8w + 255) / 256;
  cvt_bf16_kernel<<<gx, 256, 0, stream>>>(q, xqb, n8x);
  cvt_bf16_kernel<<<gx, 256, 0, stream>>>(k, xkb, n8x);
  cvt_bf16_kernel<<<gx, 256, 0, stream>>>(v, xvb, n8x);
  cvt_bf16_kernel<<<gw, 256, 0, stream>>>(Wq, wqb, n8w);
  cvt_bf16_kernel<<<gw, 256, 0, stream>>>(Wk, wkb, n8w);
  cvt_bf16_kernel<<<gw, 256, 0, stream>>>(Wv, wvb, n8w);
  cvt_bf16_kernel<<<gw, 256, 0, stream>>>(Wo, wob, n8w);

  // 1) QKV projections (z selects Q/K/V), head-split bf16 output.
  qkv_proj_kernel<<<dim3(M / 128, D_MODEL / 64, 3), 128, 0, stream>>>(
      xqb, xkb, xvb, wqb, bq, wkb, bk, wvb, bv, qh, kh, vh);

  // 2) Flash attention over 24 (b,h) pairs, 64-row Q blocks.
  flash_attn_kernel<<<dim3(S_LEN / 64, B_BATCH * HEADS), 128, 0, stream>>>(
      qh, kh, vh, ctx);

  // 3) Output projection to fp32.
  out_proj_kernel<<<dim3(M / 128, D_MODEL / 64), 128, 0, stream>>>(
      ctx, wob, bo, out);
}
